// NghSampler_9921374453827
// MI455X (gfx1250) — matrix operand order repaired and checked by reading, not compiled, not run on hardware
//
#include <hip/hip_runtime.h>

typedef __attribute__((ext_vector_type(16))) _Float16 v16h;
typedef __attribute__((ext_vector_type(8)))  _Float16 v8h;
typedef __attribute__((ext_vector_type(8)))  float    v8f;

namespace ngh {
constexpr int B  = 4;
constexpr int D  = 128;
constexpr int H  = 192;
constexpr int W  = 192;
constexpr int BORD = 8;
constexpr int Hq = H - 2 * BORD;   // 176
constexpr int Wq = W - 2 * BORD;   // 176
constexpr int NCH = 145;           // 1 + 144 disc offsets
constexpr size_t HW   = (size_t)H * W;
constexpr size_t SC_N = (size_t)B * Hq * Wq * NCH;
constexpr size_t QN   = (size_t)B * Hq * Wq;
// Sink for masked-off scatter stores: first element of the gt region.
// ngh_aux_kernel runs AFTER the correlation kernel and rewrites all of gt,
// so the garbage landing here is deterministically overwritten.
constexpr unsigned SINK = (unsigned)SC_N;
}

// grid_sample coordinate dance from the reference (align_corners=False path).
__device__ inline void ngh_sample_coords(const float* __restrict__ aflow,
                                         int b, int pix, float& xs, float& ys) {
    using namespace ngh;
    float ax = aflow[((size_t)b * 2 + 0) * HW + pix];
    float ay = aflow[((size_t)b * 2 + 1) * HW + pix];
    float gx = ax * (2.0f / (W - 1)) - 1.0f;
    float gy = ay * (2.0f / (H - 1)) - 1.0f;
    if (gx != gx) gx = 9e9f;
    if (gy != gy) gy = 9e9f;
    xs = ((gx + 1.0f) * W - 1.0f) * 0.5f;
    ys = ((gy + 1.0f) * H - 1.0f) * 0.5f;
}

// ---------------------------------------------------------------------------
// Phase 1: bilinear-warp feat2 -> channels-last f16; feat1 -> channels-last f16
// Both tensors are ~37.7 MB: fully resident in the MI455X's 192 MB L2, so the
// correlation kernel's heavy window reuse is served from L2 (no LDS staging).
// ---------------------------------------------------------------------------
__global__ __launch_bounds__(256) void ngh_warp_cvt_kernel(
    const float* __restrict__ feat1, const float* __restrict__ feat2,
    const float* __restrict__ aflow,
    _Float16* __restrict__ f1h, _Float16* __restrict__ f2h) {
    using namespace ngh;
    int p = blockIdx.x * blockDim.x + threadIdx.x;
    if (p >= B * H * W) return;
    int b   = p / (int)HW;
    int pix = p - b * (int)HW;

    float xs, ys;
    ngh_sample_coords(aflow, b, pix, xs, ys);
    float xf = floorf(xs), yf = floorf(ys);
    float fx1 = xs - xf, fx0 = 1.0f - fx1;
    float fy1 = ys - yf, fy0 = 1.0f - fy1;
    bool vx0 = (xf >= 0.0f) && (xf <= (float)(W - 1));
    bool vx1 = (xf + 1.0f >= 0.0f) && (xf + 1.0f <= (float)(W - 1));
    bool vy0 = (yf >= 0.0f) && (yf <= (float)(H - 1));
    bool vy1 = (yf + 1.0f >= 0.0f) && (yf + 1.0f <= (float)(H - 1));
    float w00 = fx0 * fy0 * (vx0 && vy0 ? 1.0f : 0.0f);
    float w01 = fx1 * fy0 * (vx1 && vy0 ? 1.0f : 0.0f);
    float w10 = fx0 * fy1 * (vx0 && vy1 ? 1.0f : 0.0f);
    float w11 = fx1 * fy1 * (vx1 && vy1 ? 1.0f : 0.0f);
    int ix0 = (int)xf, iy0 = (int)yf;
    int cx0 = ix0 < 0 ? 0 : (ix0 > W - 1 ? W - 1 : ix0);
    int cy0 = iy0 < 0 ? 0 : (iy0 > H - 1 ? H - 1 : iy0);
    int ix1 = ix0 + 1, iy1 = iy0 + 1;
    int cx1 = ix1 < 0 ? 0 : (ix1 > W - 1 ? W - 1 : ix1);
    int cy1 = iy1 < 0 ? 0 : (iy1 > H - 1 ? H - 1 : iy1);

    size_t fbase = (size_t)b * D * HW;
    size_t o00 = (size_t)cy0 * W + cx0, o01 = (size_t)cy0 * W + cx1;
    size_t o10 = (size_t)cy1 * W + cx0, o11 = (size_t)cy1 * W + cx1;

    // warped feat2 -> f16, channels last
    for (int cc = 0; cc < D; cc += 16) {
        _Float16 acc[16];
        #pragma unroll
        for (int u = 0; u < 16; ++u) {
            size_t co = fbase + (size_t)(cc + u) * HW;
            float v = w00 * feat2[co + o00] + w01 * feat2[co + o01]
                    + w10 * feat2[co + o10] + w11 * feat2[co + o11];
            acc[u] = (_Float16)v;
        }
        *(v16h*)(f2h + (size_t)p * D + cc) = *(const v16h*)acc;
    }
    // feat1 -> f16, channels last
    for (int cc = 0; cc < D; cc += 16) {
        _Float16 acc[16];
        #pragma unroll
        for (int u = 0; u < 16; ++u)
            acc[u] = (_Float16)feat1[fbase + (size_t)(cc + u) * HW + pix];
        *(v16h*)(f1h + (size_t)p * D + cc) = *(const v16h*)acc;
    }
}

// ---------------------------------------------------------------------------
// Small outputs: gt (one-hot ch 0), mask2 (nearest in-bounds), qconf.
// Runs AFTER the correlation kernel: rewriting gt also scrubs the scatter sink.
// ---------------------------------------------------------------------------
__global__ __launch_bounds__(256) void ngh_aux_kernel(
    const float* __restrict__ conf1, const float* __restrict__ conf2,
    const float* __restrict__ aflow, float* __restrict__ out) {
    using namespace ngh;
    int q = blockIdx.x * blockDim.x + threadIdx.x;
    if (q >= (int)QN) return;
    int b  = q / (Hq * Wq);
    int r  = q - b * Hq * Wq;
    int hq = r / Wq, wq = r - hq * Wq;
    int pix = (BORD + hq) * W + (BORD + wq);

    float xs, ys;
    ngh_sample_coords(aflow, b, pix, xs, ys);

    // bilinear conf2 (zeros padding)
    float xf = floorf(xs), yf = floorf(ys);
    float fx1 = xs - xf, fx0 = 1.0f - fx1;
    float fy1 = ys - yf, fy0 = 1.0f - fy1;
    float c2 = 0.0f;
    #pragma unroll
    for (int dy = 0; dy < 2; ++dy)
        #pragma unroll
        for (int dx = 0; dx < 2; ++dx) {
            float xi = xf + dx, yi = yf + dy;
            bool v = (xi >= 0.0f) && (xi <= (float)(W - 1)) &&
                     (yi >= 0.0f) && (yi <= (float)(H - 1));
            int cx = (int)xi; cx = cx < 0 ? 0 : (cx > W - 1 ? W - 1 : cx);
            int cy = (int)yi; cy = cy < 0 ? 0 : (cy > H - 1 ? H - 1 : cy);
            float w = (dx ? fx1 : fx0) * (dy ? fy1 : fy0);
            c2 += (v ? w : 0.0f) * conf2[(size_t)b * HW + (size_t)cy * W + cx];
        }
    // nearest in-bounds mask
    float xn = floorf(xs + 0.5f), yn = floorf(ys + 0.5f);
    float msk = ((xn >= 0.0f) && (xn <= (float)(W - 1)) &&
                 (yn >= 0.0f) && (yn <= (float)(H - 1))) ? 1.0f : 0.0f;

    float* gt    = out + SC_N;
    float* mask2 = out + 2 * SC_N;
    float* qconf = mask2 + QN;
    mask2[q] = msk;
    qconf[q] = 0.5f * (conf1[(size_t)b * HW + pix] + c2);
    float* g = gt + (size_t)q * NCH;
    g[0] = 1.0f;
    for (int k = 1; k < NCH; ++k) g[k] = 0.0f;
}

// ---------------------------------------------------------------------------
// Phase 2: WMMA correlation. One wave per 16-wide query strip.
// A (16x128): 16 query pixels' f16 features. For each offset row j (fully
// unrolled), GEMM against a 32-column target window (two 16-col N-tiles),
// K in 4 steps of 32 -> 8 x v_wmma_f32_16x16x32_f16 per j (120 per strip).
// Scatter is fully branchless: invalid diagonal slots are steered to a sink
// index inside the gt region (rewritten by ngh_aux_kernel afterwards).
// ---------------------------------------------------------------------------
__global__ __launch_bounds__(256) void ngh_corr_kernel(
    const _Float16* __restrict__ f1h, const _Float16* __restrict__ f2h,
    float* __restrict__ scores) {
    using namespace ngh;
    constexpr int SW = Wq / 16;                 // 11 strips per row
    int wave = (int)((blockIdx.x * blockDim.x + threadIdx.x) >> 5);
    if (wave >= B * Hq * SW) return;            // grid sized exactly; never taken
    int lane = threadIdx.x & 31;
    int lg = lane >> 4;                          // lane group (K-half select)
    int n  = lane & 15;                          // N index / A row index

    int b  = wave / (Hq * SW);
    int r  = wave - b * Hq * SW;
    int hq = r / SW;
    int w0 = (r - hq * SW) * 16;
    int y  = BORD + hq;
    int x0 = BORD + w0;

    // --- A fragments: row m = n, K slices contiguous (k = 16*lg + s), same order as B ---
    v16h A[4];
    {
        const _Float16* qp = f1h + (((size_t)(b * H + y)) * W + (x0 + n)) * D;
        #pragma unroll
        for (int kk = 0; kk < 4; ++kk) {
            int bk = kk * 32 + lg * 16;
            A[kk] = *(const v16h*)(qp + bk);
        }
    }

    // --- j-invariant B base pointers (row y; j adds an immediate offset) ---
    int xc0 = x0 - 7 + n;                        // window cols [x0-7 .. x0+8]
    int xc1 = x0 + 9 + n;                        // window cols [x0+9 .. x0+24]
    if (xc1 > W - 1) xc1 = W - 1;                // pad cols (never scattered)
    const _Float16* bbase0 =
        f2h + (((size_t)(b * H + y)) * W + xc0) * D + lg * 16;
    const _Float16* bbase1 =
        f2h + (((size_t)(b * H + y)) * W + xc1) * D + lg * 16;

    // --- j-invariant scatter state per (tile, v) element ------------------
    // i    = diagonal offset (column offset of this C element)
    // u    = i + 7 (validity tests become unsigned range checks)
    // qi2  = element index of scores[b,hq,w0+m, i]  (i.e. base chan + i)
    int u_[16];
    unsigned qi2[16];
    #pragma unroll
    for (int e = 0; e < 16; ++e) {
        int m = (e & 7) + lg * 8;
        int i = (e < 8) ? (n - 7 - m) : (n + 9 - m);
        u_[e]  = i + 7;
        qi2[e] = (unsigned)((((b * Hq + hq)) * Wq + (w0 + m)) * NCH + i);
    }

    constexpr int IMAX[8]   = {7, 6, 6, 6, 5, 4, 3, 0};   // by |j|
    constexpr int START[15] = {1, 2, 9, 18, 29, 42, 55, 67,
                               79, 91, 104, 117, 128, 137, 144};

    #pragma unroll
    for (int j = -7; j <= 7; ++j) {
        const int jm   = j < 0 ? -j : j;
        const int imax = IMAX[jm];
        const int sb   = START[j + 7] + imax;    // chan = sb + i (hole-adjusted)
        const long rowo = (long)j * W * D;       // folds into load offsets

        v8f c0 = {}; v8f c1 = {};
        #pragma unroll
        for (int kk = 0; kk < 4; ++kk) {
            v16h b0 = *(const v16h*)(bbase0 + rowo + kk * 32);
            v16h b1 = *(const v16h*)(bbase1 + rowo + kk * 32);
            c0 = __builtin_amdgcn_wmma_f32_16x16x32_f16(
                     false, A[kk], false, b0, (short)0, c0, false, false);
            c1 = __builtin_amdgcn_wmma_f32_16x16x32_f16(
                     false, A[kk], false, b1, (short)0, c1, false, false);
        }

        #pragma unroll
        for (int e = 0; e < 16; ++e) {
            float cval = (e < 8) ? c0[e & 7] : c1[e & 7];
            unsigned idx;
            bool ok;
            if (jm >= 2) {
                // chan = sb + i; valid iff |i| <= imax
                ok  = (unsigned)(u_[e] + (imax - 7)) <= (unsigned)(2 * imax);
                idx = qi2[e] + (unsigned)sb;
            } else if (jm == 1) {
                // exclude i==0; channels shift by -1 for i>0
                int i = u_[e] - 7;
                ok  = ((unsigned)(u_[e] + (imax - 7)) <= (unsigned)(2 * imax))
                      && (i != 0);
                idx = qi2[e] + (unsigned)(i > 0 ? sb - 1 : sb);
            } else {
                // j == 0: exclude |i|==1; i==0 -> channel 0; -3 shift for i>0
                int i = u_[e] - 7;
                ok  = ((unsigned)u_[e] <= 14u) && (i != 1) && (i != -1);
                int add = (i == 0) ? 0 : (i > 0 ? sb - 3 : sb);  // sb==74
                idx = qi2[e] + (unsigned)add;
            }
            scores[ok ? idx : SINK] = cval;      // branchless: cndmask + store
        }
    }
}

extern "C" void kernel_launch(void* const* d_in, const int* in_sizes, int n_in,
                              void* d_out, int out_size, void* d_ws, size_t ws_size,
                              hipStream_t stream) {
    using namespace ngh;
    const float* feat1 = (const float*)d_in[0];
    const float* feat2 = (const float*)d_in[1];
    const float* conf1 = (const float*)d_in[2];
    const float* conf2 = (const float*)d_in[3];
    const float* aflow = (const float*)d_in[4];
    float* out = (float*)d_out;

    _Float16* f1h = (_Float16*)d_ws;                 // 37.75 MB
    _Float16* f2h = f1h + (size_t)B * HW * D;        // 37.75 MB

    int npix = B * H * W;                            // 147456 -> 576 blocks
    ngh_warp_cvt_kernel<<<(npix + 255) / 256, 256, 0, stream>>>(
        feat1, feat2, aflow, f1h, f2h);

    int waves = B * Hq * (Wq / 16);                  // 7744 waves, 8 per block
    ngh_corr_kernel<<<(waves + 7) / 8, 256, 0, stream>>>(f1h, f2h, out);

    // After corr: rewrites the whole gt region, scrubbing the scatter sink.
    ngh_aux_kernel<<<((int)QN + 255) / 256, 256, 0, stream>>>(
        conf1, conf2, aflow, out);
}
